// Block_87213605913025
// MI455X (gfx1250) — matrix operand order repaired and checked by reading, not compiled, not run on hardware
//
#include <hip/hip_runtime.h>

// ---------------------------------------------------------------------------
// Transformer block forward (eval) for gfx1250 / MI455X, wave32 + WMMA.
// Matmuls on v_wmma_f32_16x16x32_f16; tile staging via async-to-LDS copies.
// Weights are cast f32->f16 *pre-transposed* so staging is contiguous.
// GEMM: 256x64 block tile, 32 rows/wave -> 8 WMMAs per 8 ds_load_b128.
// ---------------------------------------------------------------------------

typedef _Float16 f16;
typedef __attribute__((ext_vector_type(16))) _Float16 v16h;
typedef __attribute__((ext_vector_type(8)))  _Float16 v8h;
typedef __attribute__((ext_vector_type(8)))  float    v8f;

#define N_EMBED 384
#define T_SEQ   256
#define BATCH   128
#define NTOK    (BATCH * T_SEQ)   // 32768 tokens
#define QKV_LD  1152              // q(384) | k(384) | v(384)
#define FF_DIM  1536

__device__ __forceinline__ v8f wmma_f16f32(v16h a, v16h b, v8f c) {
  // D = A(16x32 f16) * B(32x16 f16) + C(16x16 f32)
  return __builtin_amdgcn_wmma_f32_16x16x32_f16(
      /*neg_a=*/false, a, /*neg_b=*/false, b,
      /*c_mod=*/(short)0, c, /*reuse_a=*/false, /*reuse_b=*/false);
}

__device__ __forceinline__ v16h cat8(v8h a, v8h b) {
  v16h r;
#pragma unroll
  for (int i = 0; i < 8; ++i) { r[i] = a[i]; r[i + 8] = b[i]; }
  return r;
}

__device__ __forceinline__ v8f zero8() {
  v8f z;
#pragma unroll
  for (int i = 0; i < 8; ++i) z[i] = 0.0f;
  return z;
}

// Async copy of 16 bytes/lane from global to LDS (ASYNCcnt-tracked, no VGPRs).
__device__ __forceinline__ void async_copy_b128(unsigned lds_off, const void* g) {
  asm volatile("global_load_async_to_lds_b128 %0, %1, off"
               :: "v"(lds_off), "v"((unsigned long long)(uintptr_t)g)
               : "memory");
}
__device__ __forceinline__ void wait_async0() {
  asm volatile("s_wait_asynccnt 0" ::: "memory");
}
__device__ __forceinline__ unsigned lds_off_of(const void* p) {
  return (unsigned)(uintptr_t)p;   // generic LDS addr: low 32 bits = LDS offset
}

// ---------------------------------------------------------------------------
// Weight casts (f32 -> f16), stored TRANSPOSED: dst[N][K] (n = output col).
// wq/wk/wv [H=6][384][64] fuse into dst[1152][384].
// ---------------------------------------------------------------------------
__global__ void cast_wqkv_kernel(const float* __restrict__ wq,
                                 const float* __restrict__ wk,
                                 const float* __restrict__ wv,
                                 f16* __restrict__ dst) {
  int i = blockIdx.x * blockDim.x + threadIdx.x;
  if (i >= QKV_LD * N_EMBED) return;
  int n = i / N_EMBED, c = i % N_EMBED;       // n: output col, c: input dim (K)
  int sel = n / N_EMBED, jj = n % N_EMBED;
  int h = jj >> 6, d = jj & 63;
  const float* src = (sel == 0) ? wq : (sel == 1) ? wk : wv;
  dst[i] = (f16)src[((size_t)h * N_EMBED + c) * 64 + d];
}

// src is [K][N] row-major f32; dst is [N][K] f16.
__global__ void cast_f16T_kernel(const float* __restrict__ src,
                                 f16* __restrict__ dst, int K, int N) {
  int i = blockIdx.x * blockDim.x + threadIdx.x;
  if (i >= K * N) return;
  int n = i / K, k = i % K;
  dst[i] = (f16)src[(size_t)k * N + n];
}

// ---------------------------------------------------------------------------
// LayerNorm over last dim (384), f32 in -> f16 out. 128 threads per row.
// ---------------------------------------------------------------------------
__global__ void ln_f16_kernel(const float* __restrict__ x,
                              const float* __restrict__ g,
                              const float* __restrict__ b,
                              f16* __restrict__ out) {
  const int row = blockIdx.x;
  const int tid = threadIdx.x;               // 0..127
  const float* xr = x + (size_t)row * N_EMBED;
  float v0 = xr[tid], v1 = xr[tid + 128], v2 = xr[tid + 256];
  float s = v0 + v1 + v2;
  float q = v0 * v0 + v1 * v1 + v2 * v2;
#pragma unroll
  for (int off = 16; off > 0; off >>= 1) {
    s += __shfl_xor(s, off, 32);
    q += __shfl_xor(q, off, 32);
  }
  __shared__ float rs[4], rq[4];
  const int w = tid >> 5, l = tid & 31;
  if (l == 0) { rs[w] = s; rq[w] = q; }
  __syncthreads();
  const float S = rs[0] + rs[1] + rs[2] + rs[3];
  const float Q = rq[0] + rq[1] + rq[2] + rq[3];
  const float mean = S * (1.0f / N_EMBED);
  const float var  = Q * (1.0f / N_EMBED) - mean * mean;
  const float rstd = rsqrtf(var + 1e-5f);
  f16* orow = out + (size_t)row * N_EMBED;
  orow[tid]       = (f16)((v0 - mean) * rstd * g[tid]       + b[tid]);
  orow[tid + 128] = (f16)((v1 - mean) * rstd * g[tid + 128] + b[tid + 128]);
  orow[tid + 256] = (f16)((v2 - mean) * rstd * g[tid + 256] + b[tid + 256]);
}

// ---------------------------------------------------------------------------
// WMMA GEMM: C[M,N] = A[M,K](f16) * BT[N,K](f16, pre-transposed weights)
// (+bias)(+res)(relu). 256 threads = 8 waves; block tile 256(M) x 64(N),
// 32 rows per wave (two 16-row sub-tiles share the B fragments), K-step 32.
// B tile double-buffered in LDS, filled by async-to-LDS copies overlapped
// with WMMA compute. M%256==0, N%64==0, K%32==0 assumed.
// ---------------------------------------------------------------------------
__global__ void gemm_wmma_kernel(const f16* __restrict__ A,
                                 const f16* __restrict__ BT,
                                 float* out32, f16* out16,
                                 const float* __restrict__ bias,
                                 const float* __restrict__ res,
                                 int M, int N, int K, int relu) {
  __shared__ alignas(16) f16 bt[2][64 * 32];   // [buf][n][k]
  const int tid  = threadIdx.x;
  const int wave = tid >> 5;
  const int lane = tid & 31;
  const int lh   = lane >> 4;                // half of wave
  const int ln   = lane & 15;
  const int rowBase = blockIdx.y * 256 + wave * 32;
  const int n0      = blockIdx.x * 64;

  v8f acc[2][4];
#pragma unroll
  for (int rs = 0; rs < 2; ++rs)
#pragma unroll
    for (int t = 0; t < 4; ++t) acc[rs][t] = zero8();

  const f16* arow0 = A + (size_t)(rowBase + ln) * K;
  const f16* arow1 = A + (size_t)(rowBase + 16 + ln) * K;

  // staging: thread -> (n within tile, k chunk); contiguous 16B per thread
  const int nn = tid >> 2;                    // 0..63
  const int kc = (tid & 3) * 8;               // 0,8,16,24
  const f16* gB = BT + (size_t)(n0 + nn) * K + kc;
  const unsigned ldsB0 = lds_off_of(&bt[0][nn * 32 + kc]);
  const unsigned ldsB1 = lds_off_of(&bt[1][nn * 32 + kc]);

  async_copy_b128(ldsB0, gB);                 // prologue: tile 0 in flight
  int buf = 0;
  for (int k0 = 0; k0 < K; k0 += 32) {
    wait_async0();                            // this wave's current tile done
    __syncthreads();                          // all waves' parts done
    if (k0 + 32 < K)                          // overlap next copy with compute
      async_copy_b128(buf ? ldsB0 : ldsB1, gB + k0 + 32);

    // A fragments: lanes 0-15 rows m=ln K{0..15}; lanes 16-31 K{16..31}
    v16h af0 = cat8(*(const v8h*)(arow0 + k0 + lh * 16),
                    *(const v8h*)(arow0 + k0 + 8 + lh * 16));
    v16h af1 = cat8(*(const v8h*)(arow1 + k0 + lh * 16),
                    *(const v8h*)(arow1 + k0 + 8 + lh * 16));

    const f16* btb = &bt[buf][0];
#pragma unroll
    for (int t = 0; t < 4; ++t) {
      v16h bf = *(const v16h*)&btb[(t * 16 + ln) * 32 + lh * 16];
      acc[0][t] = wmma_f16f32(af0, bf, acc[0][t]);
      acc[1][t] = wmma_f16f32(af1, bf, acc[1][t]);
    }
    buf ^= 1;
  }

  // Epilogue. C layout: VGPR r -> row r + 8*lh, col = ln (within 16-tile).
#pragma unroll
  for (int rs = 0; rs < 2; ++rs) {
#pragma unroll
    for (int t = 0; t < 4; ++t) {
      const int gn = n0 + t * 16 + ln;
      const float bv = bias ? bias[gn] : 0.0f;
#pragma unroll
      for (int r = 0; r < 8; ++r) {
        const int gm = rowBase + rs * 16 + r + 8 * lh;
        float v = acc[rs][t][r] + bv;
        if (res)  v += res[(size_t)gm * N + gn];
        if (relu) v = fmaxf(v, 0.0f);
        if (out32) out32[(size_t)gm * N + gn] = v;
        else       out16[(size_t)gm * N + gn] = (f16)v;
      }
    }
  }
}

// ---------------------------------------------------------------------------
// Flash-style causal attention. grid = (2 rowblocks, 6 heads, 128 batches),
// 256 threads (8 waves x 16 query rows). Key tiles of 32 streamed via LDS
// (K tile async-copied; V tile transposed manually).
// scale = N_EMBED^-0.5 (matches reference, NOT head_size^-0.5).
// ---------------------------------------------------------------------------
__global__ void attn_kernel(const f16* __restrict__ qkv, f16* __restrict__ o16) {
  const int blk = blockIdx.x;      // 0..1  -> query rows [blk*128, +128)
  const int h   = blockIdx.y;      // 0..5
  const int b   = blockIdx.z;      // 0..127
  const int tid  = threadIdx.x;
  const int wave = tid >> 5;
  const int lane = tid & 31;
  const int lh   = lane >> 4;
  const int ln   = lane & 15;
  const int qr0  = blk * 128 + wave * 16;     // first query row of this wave
  const int tokBase = b * T_SEQ;
  const float SCALE = 0.051031036307982884f;  // 384^-0.5

  const f16* qptr = qkv + (size_t)tokBase * QKV_LD + h * 64;
  const f16* kptr = qptr + 384;
  const f16* vptr = qptr + 768;

  __shared__ alignas(16) f16 lds_k[32 * 64];      // [j][d]
  __shared__ alignas(16) f16 lds_vT[64 * 32];     // [d][j]
  __shared__ alignas(16) f16 lds_p[8][16 * 32];   // per-wave P staging

  // Q fragments for this wave's 16 rows (K-dim = 64 -> two 16x32 A frags)
  v16h aq0, aq1;
  {
    const f16* qrow = qptr + (size_t)(qr0 + ln) * QKV_LD;
    v8h a0 = *(const v8h*)(qrow + lh * 16);
    v8h a1 = *(const v8h*)(qrow + 8 + lh * 16);
    v8h a2 = *(const v8h*)(qrow + 32 + lh * 16);
    v8h a3 = *(const v8h*)(qrow + 40 + lh * 16);
    aq0 = cat8(a0, a1);
    aq1 = cat8(a2, a3);
  }

  float rowM[8], rowS[8];
  v8f oacc[4];
#pragma unroll
  for (int r = 0; r < 8; ++r) { rowM[r] = -INFINITY; rowS[r] = 0.0f; }
#pragma unroll
  for (int t = 0; t < 4; ++t) oacc[t] = zero8();

  const int nTiles = (blk + 1) * 4;           // causal extent for this block
  const int cr = tid >> 3;                    // coop-load row 0..31
  const int cc = (tid & 7) * 8;               // coop-load col start
  const unsigned ldsKoff = lds_off_of(&lds_k[cr * 64 + cc]);

  for (int tile = 0; tile < nTiles; ++tile) {
    const int j0 = tile * 32;
    // --- stage K tile via async copy; V tile transposed manually ---
    async_copy_b128(ldsKoff, kptr + (size_t)(j0 + cr) * QKV_LD + cc);
    {
      v8h gv = *(const v8h*)(vptr + (size_t)(j0 + cr) * QKV_LD + cc);
#pragma unroll
      for (int i = 0; i < 8; ++i) lds_vT[(cc + i) * 32 + cr] = gv[i];
    }
    wait_async0();
    __syncthreads();

    if (j0 <= qr0 + 15) {   // wave-uniform: this wave has unmasked keys here
      // --- scores S = Q K^T (two 16-key subtiles, K-dim 64 chained) ---
      v8f s[2];
#pragma unroll
      for (int t = 0; t < 2; ++t) {
        v16h bk0 = *(const v16h*)&lds_k[(t * 16 + ln) * 64 + 0 + lh * 16];
        v16h bk1 = *(const v16h*)&lds_k[(t * 16 + ln) * 64 + 32 + lh * 16];
        v8f sa = wmma_f16f32(aq0, bk0, zero8());
        s[t]   = wmma_f16f32(aq1, bk1, sa);
      }
      // --- scale + causal mask + online softmax ---
      float pf[2][8];
#pragma unroll
      for (int r = 0; r < 8; ++r) {
        const int qabs = qr0 + r + 8 * lh;
        float sc0 = s[0][r] * SCALE;
        float sc1 = s[1][r] * SCALE;
        if (j0 + ln > qabs)      sc0 = -INFINITY;
        if (j0 + 16 + ln > qabs) sc1 = -INFINITY;
        float mx = fmaxf(sc0, sc1);
#pragma unroll
        for (int off = 8; off > 0; off >>= 1) mx = fmaxf(mx, __shfl_xor(mx, off, 16));
        const float newM  = fmaxf(rowM[r], mx);
        const float alpha = __expf(rowM[r] - newM);
        const float p0 = __expf(sc0 - newM);
        const float p1 = __expf(sc1 - newM);
        float ps = p0 + p1;
#pragma unroll
        for (int off = 8; off > 0; off >>= 1) ps += __shfl_xor(ps, off, 16);
        rowS[r] = rowS[r] * alpha + ps;
        rowM[r] = newM;
#pragma unroll
        for (int t = 0; t < 4; ++t) oacc[t][r] *= alpha;
        pf[0][r] = p0; pf[1][r] = p1;
      }
      // --- P: C-layout -> A-layout via per-wave LDS round trip ---
      f16* myP = &lds_p[wave][0];
#pragma unroll
      for (int t = 0; t < 2; ++t)
#pragma unroll
        for (int r = 0; r < 8; ++r)
          myP[(r + 8 * lh) * 32 + t * 16 + ln] = (f16)pf[t][r];
      asm volatile("s_wait_dscnt 0" ::: "memory");
      v8h p0 = *(const v8h*)&myP[ln * 32 + lh * 16];
      v8h p1 = *(const v8h*)&myP[ln * 32 + 8 + lh * 16];
      v16h ap = cat8(p0, p1);
      // --- O += P @ V (4 d-subtiles) ---
#pragma unroll
      for (int t = 0; t < 4; ++t) {
        v16h bv = *(const v16h*)&lds_vT[(t * 16 + ln) * 32 + lh * 16];
        oacc[t] = wmma_f16f32(ap, bv, oacc[t]);
      }
    }
    __syncthreads();
  }

  // --- normalize and write O (heads concatenated: cols h*64 .. h*64+64) ---
#pragma unroll
  for (int t = 0; t < 4; ++t) {
#pragma unroll
    for (int r = 0; r < 8; ++r) {
      const float v = oacc[t][r] / rowS[r];
      const size_t tok = (size_t)tokBase + qr0 + r + 8 * lh;
      o16[tok * N_EMBED + h * 64 + t * 16 + ln] = (f16)v;
    }
  }
}

// ---------------------------------------------------------------------------
// Host-side orchestration. Workspace layout (bytes, all 256-aligned):
//   [0,            25165824)  h16   (reused later as h2_16)
//   [25165824,    100663296)  qkv16          \ reused together as
//   [100663296,   125829120)  o16            / ff16 (100663296 B)
//   [125829120,   129368064)  f16 weights (wqkvT | wprojT | w1T | w2T)
// Total: ~123.4 MiB.
// ---------------------------------------------------------------------------
extern "C" void kernel_launch(void* const* d_in, const int* in_sizes, int n_in,
                              void* d_out, int out_size, void* d_ws, size_t ws_size,
                              hipStream_t stream) {
  (void)in_sizes; (void)n_in; (void)out_size; (void)ws_size;
  const float* x      = (const float*)d_in[0];
  const float* wq     = (const float*)d_in[1];
  const float* wk     = (const float*)d_in[2];
  const float* wv     = (const float*)d_in[3];
  const float* w_proj = (const float*)d_in[4];
  const float* b_proj = (const float*)d_in[5];
  const float* w1     = (const float*)d_in[6];
  const float* b1     = (const float*)d_in[7];
  const float* w2     = (const float*)d_in[8];
  const float* b2     = (const float*)d_in[9];
  const float* ln1_g  = (const float*)d_in[10];
  const float* ln1_b  = (const float*)d_in[11];
  const float* ln2_g  = (const float*)d_in[12];
  const float* ln2_b  = (const float*)d_in[13];
  float* out = (float*)d_out;

  char* ws = (char*)d_ws;
  f16* h16     = (f16*)(ws + 0);
  f16* qkv16   = (f16*)(ws + 25165824u);
  f16* o16     = (f16*)(ws + 100663296u);
  f16* ff16    = (f16*)(ws + 25165824u);     // overlays dead qkv16+o16
  f16* wqkvT   = (f16*)(ws + 125829120u);    // [1152][384]
  f16* wprojT  = (f16*)(ws + 126713856u);    // [384][384]
  f16* w1T     = (f16*)(ws + 127008768u);    // [1536][384]
  f16* w2T     = (f16*)(ws + 128188416u);    // [384][1536]

  // 1) weight casts (transposed)
  cast_wqkv_kernel<<<(QKV_LD * N_EMBED + 255) / 256, 256, 0, stream>>>(wq, wk, wv, wqkvT);
  cast_f16T_kernel<<<(N_EMBED * N_EMBED + 255) / 256, 256, 0, stream>>>(w_proj, wprojT, N_EMBED, N_EMBED);
  cast_f16T_kernel<<<(N_EMBED * FF_DIM + 255) / 256, 256, 0, stream>>>(w1, w1T, N_EMBED, FF_DIM);
  cast_f16T_kernel<<<(FF_DIM * N_EMBED + 255) / 256, 256, 0, stream>>>(w2, w2T, FF_DIM, N_EMBED);

  // 2) h = LN1(x) -> f16
  ln_f16_kernel<<<NTOK, 128, 0, stream>>>(x, ln1_g, ln1_b, h16);

  // 3) qkv = h @ Wqkv   [32768,384]x[384,1152] -> f16
  gemm_wmma_kernel<<<dim3(QKV_LD / 64, NTOK / 256), 256, 0, stream>>>(
      h16, wqkvT, nullptr, qkv16, nullptr, nullptr, NTOK, QKV_LD, N_EMBED, 0);

  // 4) causal attention -> o16 [32768,384]
  attn_kernel<<<dim3(2, 6, BATCH), 256, 0, stream>>>(qkv16, o16);

  // 5) x1 = x + o @ Wproj + b_proj -> d_out (f32)
  gemm_wmma_kernel<<<dim3(N_EMBED / 64, NTOK / 256), 256, 0, stream>>>(
      o16, wprojT, out, nullptr, b_proj, x, NTOK, N_EMBED, N_EMBED, 0);

  // 6) h2 = LN2(x1) -> f16 (reuse h16 region)
  ln_f16_kernel<<<NTOK, 128, 0, stream>>>(out, ln2_g, ln2_b, h16);

  // 7) ff = relu(h2 @ W1 + b1) -> f16
  gemm_wmma_kernel<<<dim3(FF_DIM / 64, NTOK / 256), 256, 0, stream>>>(
      h16, w1T, nullptr, ff16, b1, nullptr, NTOK, FF_DIM, N_EMBED, 1);

  // 8) out = x1 + ff @ W2 + b2 (in-place residual on d_out)
  gemm_wmma_kernel<<<dim3(N_EMBED / 64, NTOK / 256), 256, 0, stream>>>(
      ff16, w2T, out, nullptr, b2, out, NTOK, N_EMBED, FF_DIM, 0);
}
